// LSTM_17463337025754
// MI455X (gfx1250) — compile-verified, matrix operand order checked by reading, not run on hardware
//
#include <hip/hip_runtime.h>
#include <cstdint>
#include <cstddef>

// ---------------------------------------------------------------------------
// LSTM forward for MI455X (gfx1250, wave32, WMMA).
//
// Phase 0 (parallel prep):   x -> bf16 copy; W^T = [Wi;Wh]^T -> bf16 [2048,1024];
//                            bias = bi+bh; h/c state init; barrier ctl init.
// Phase 1 (persistent scan): for t in 0..1023:
//     gates[128,2048] = [x_t | h] (bf16) @ W^T  via v_wmma_f32_16x16x32_bf16
//     grid barrier
//     c = sig(f)*c + sig(i)*tanh(g); h = sig(o)*tanh(c); emit h (+hy,cy at end)
//     grid barrier
// W^T (4 MiB bf16) stays L2-resident (192 MB L2); scan is latency-bound, so
// staging uses gfx1250 async-to-LDS copies and each WG owns 32 gate columns
// (2 WMMA tiles/wave per K-chunk) to amortize the device-wide barriers.
// ---------------------------------------------------------------------------

#define DEV __device__ __forceinline__

typedef __attribute__((ext_vector_type(16))) __bf16 v16bf;
typedef __attribute__((ext_vector_type(8)))  __bf16 v8bf;
typedef __attribute__((ext_vector_type(8)))  float  v8f;
typedef __attribute__((ext_vector_type(4)))  int    v4i;

constexpr int Bc = 128, Tt = 1024, Ii = 512, Hh = 512;
constexpr int Gg = 4 * Hh;       // 2048 gate columns
constexpr int KK = Ii + Hh;      // 1024 fused K
constexpr int NWG = 64;          // persistent workgroups (32 gate cols each)
constexpr int TPB = 256;         // 8 waves -> 8 x 16 batch rows

// ---- workspace layout (bytes) ----
constexpr size_t OFF_XBF   = 0;
constexpr size_t SZ_XBF    = (size_t)Bc * Tt * Ii * 2;     // 128 MiB  x in bf16
constexpr size_t OFF_WT    = OFF_XBF + SZ_XBF;
constexpr size_t SZ_WT     = (size_t)Gg * KK * 2;          // 4 MiB    W^T bf16 [N][K]
constexpr size_t OFF_BIAS  = OFF_WT + SZ_WT;
constexpr size_t SZ_BIAS   = (size_t)Gg * 4;               // bi+bh
constexpr size_t OFF_GATES = OFF_BIAS + SZ_BIAS;
constexpr size_t SZ_GATES  = (size_t)Bc * Gg * 4;          // 1 MiB per-step gates
constexpr size_t OFF_HBF   = OFF_GATES + SZ_GATES;
constexpr size_t SZ_HBF    = (size_t)Bc * Hh * 2;          // h state bf16
constexpr size_t OFF_C     = OFF_HBF + SZ_HBF;
constexpr size_t SZ_C      = (size_t)Bc * Hh * 4;          // c state f32
constexpr size_t OFF_CTL   = OFF_C + SZ_C;                 // 2 ints: {count, generation}

// -------- gfx1250 async-copy path (guarded; falls back to sync copies) ------
#if defined(__HIP_DEVICE_COMPILE__) && defined(__gfx1250__) && \
    __has_builtin(__builtin_amdgcn_global_load_async_to_lds_b128) && \
    __has_builtin(__builtin_amdgcn_s_wait_asynccnt)
#define USE_ASYNC 1
#else
#define USE_ASYNC 0
#endif

DEV void cp16(unsigned short* dst_lds, const unsigned short* src_glb) {
#if USE_ASYNC
  unsigned short* s_nc = const_cast<unsigned short*>(src_glb);
  __attribute__((address_space(1))) v4i* gs =
      (__attribute__((address_space(1))) v4i*)s_nc;
  __attribute__((address_space(3))) v4i* ld =
      (__attribute__((address_space(3))) v4i*)dst_lds;
  __builtin_amdgcn_global_load_async_to_lds_b128(gs, ld, 0 /*offset*/, 0 /*cpol*/);
#else
  *(v8bf*)dst_lds = *(const v8bf*)src_glb;
#endif
}

DEV void wait_async_copies() {
#if USE_ASYNC
  __builtin_amdgcn_s_wait_asynccnt(0);
#endif
}

DEV unsigned short f2bf(float f) {
  unsigned u = __float_as_uint(f);
  unsigned r = u + 0x7FFFu + ((u >> 16) & 1u);   // round-to-nearest-even
  return (unsigned short)(r >> 16);
}

DEV float sigmoid_f(float x) { return 1.0f / (1.0f + __expf(-x)); }

// Sense-reversing device-wide barrier on scratch memory.
DEV void grid_barrier(int* ctl) {
  __syncthreads();
  if (threadIdx.x == 0) {
    int* cnt = ctl;
    int* gen = ctl + 1;
    int g = __hip_atomic_load(gen, __ATOMIC_RELAXED, __HIP_MEMORY_SCOPE_AGENT);
    int prev = __hip_atomic_fetch_add(cnt, 1, __ATOMIC_ACQ_REL, __HIP_MEMORY_SCOPE_AGENT);
    if (prev == NWG - 1) {
      __hip_atomic_store(cnt, 0, __ATOMIC_RELAXED, __HIP_MEMORY_SCOPE_AGENT);
      __hip_atomic_fetch_add(gen, 1, __ATOMIC_RELEASE, __HIP_MEMORY_SCOPE_AGENT);
    } else {
      while (__hip_atomic_load(gen, __ATOMIC_ACQUIRE, __HIP_MEMORY_SCOPE_AGENT) == g) {
        __builtin_amdgcn_s_sleep(1);
      }
    }
  }
  __syncthreads();
}

// ------------------------------ prep kernels --------------------------------
__global__ void k_cvt_x(const float* __restrict__ x, unsigned short* __restrict__ xb) {
  size_t i = ((size_t)blockIdx.x * blockDim.x + threadIdx.x) * 4;
  float4 v = *(const float4*)(x + i);
  uint2 r;
  r.x = (unsigned)f2bf(v.x) | ((unsigned)f2bf(v.y) << 16);
  r.y = (unsigned)f2bf(v.z) | ((unsigned)f2bf(v.w) << 16);
  *(uint2*)(xb + i) = r;
}

// W^T[n][k] = (k < I ? Wi[k][n] : Wh[k-I][n]) in bf16 (coalesced reads).
__global__ void k_prep_w(const float* __restrict__ Wi, const float* __restrict__ Wh,
                         unsigned short* __restrict__ wt) {
  size_t idx = (size_t)blockIdx.x * blockDim.x + threadIdx.x;  // over KK*Gg
  size_t k = idx / Gg, n = idx % Gg;
  float v = (idx < (size_t)Ii * Gg) ? Wi[idx] : Wh[idx - (size_t)Ii * Gg];
  wt[n * KK + k] = f2bf(v);
}

__global__ void k_init(const float* __restrict__ h0, const float* __restrict__ c0,
                       const float* __restrict__ bi, const float* __restrict__ bh,
                       unsigned short* __restrict__ hbf, float* __restrict__ cbuf,
                       float* __restrict__ bias, int* __restrict__ ctl) {
  int i = blockIdx.x * blockDim.x + threadIdx.x;
  if (i < Bc * Hh) { hbf[i] = f2bf(h0[i]); cbuf[i] = c0[i]; }
  if (i < Gg)      { bias[i] = bi[i] + bh[i]; }
  if (i == 0)      { ctl[0] = 0; ctl[1] = 0; }
}

// --------------------------- persistent scan kernel -------------------------
__global__ __launch_bounds__(TPB, 1)
void lstm_scan(const unsigned short* __restrict__ xbf,
               const unsigned short* __restrict__ wt,
               const float* __restrict__ bias,
               float* __restrict__ gates,
               unsigned short* __restrict__ hbf,
               float* __restrict__ cbuf,
               float* __restrict__ out,
               float* __restrict__ hy,
               float* __restrict__ cy,
               int* __restrict__ ctl) {
  // LDS tiles, double buffered. Row stride 40 keeps 16B alignment + de-banks.
  __shared__ __align__(16) unsigned short tA[2][128][40];  // A: 128 rows x K32
  __shared__ __align__(16) unsigned short tB[2][32][40];   // B: 32 cols x K32 (W^T rows)

  const int tid  = threadIdx.x;
  const int wave = tid >> 5;
  const int lane = tid & 31;
  const int n0   = blockIdx.x * 32;          // this WG's 32 gate columns

  // Per-thread staging coordinates (straight-line; no copy loop).
  const int srow = tid >> 2;                 // 0..63
  const int sseg = (tid & 3) * 8;            // 0,8,16,24 (elements)

  for (int t = 0; t < Tt; ++t) {
    // ---------------- Phase A: gates = [x_t | h] @ W^T + bias ----------------
    auto stage = [&](int buf, int ck) {
      const int k0 = ck * 32;
      // Uniform source base/stride (k0 is wave-uniform): no per-lane select.
      const unsigned short* abase;
      size_t astride;
      if (k0 < Ii) { abase = xbf + (size_t)t * Ii + k0; astride = (size_t)Tt * Ii; }
      else         { abase = hbf + (size_t)(k0 - Ii);   astride = (size_t)Hh; }
      // A: 128 rows x 4 x 16B -> two straight-line copies per thread.
      cp16(&tA[buf][srow][sseg],      abase + (size_t)srow * astride + sseg);
      cp16(&tA[buf][srow + 64][sseg], abase + (size_t)(srow + 64) * astride + sseg);
      // B: 32 rows x 4 x 16B -> threads 0..127.
      if (tid < 128) {
        cp16(&tB[buf][srow][sseg], wt + (size_t)(n0 + srow) * KK + k0 + sseg);
      }
    };

    v8f acc0 = {}, acc1 = {};
    stage(0, 0);
    wait_async_copies();
    __syncthreads();

#pragma unroll 2
    for (int ck = 0; ck < KK / 32; ++ck) {
      const int cur = ck & 1;
      if (ck + 1 < KK / 32) stage(cur ^ 1, ck + 1);

      // Per-lane fragments matching the 16-bit WMMA VGPR layouts:
      // lanes 0-15: K {0..7,16..23}; lanes 16-31: K {8..15,24..31}.
      const int mrow = wave * 16 + (lane & 15);
      const int ncl  = lane & 15;
      const int kh   = (lane >> 4) << 3;
      const v8bf a_lo  = *(const v8bf*)&tA[cur][mrow][kh];
      const v8bf a_hi  = *(const v8bf*)&tA[cur][mrow][kh + 16];
      const v8bf b0_lo = *(const v8bf*)&tB[cur][ncl][kh];
      const v8bf b0_hi = *(const v8bf*)&tB[cur][ncl][kh + 16];
      const v8bf b1_lo = *(const v8bf*)&tB[cur][16 + ncl][kh];
      const v8bf b1_hi = *(const v8bf*)&tB[cur][16 + ncl][kh + 16];
      v16bf a, b0, b1;
#pragma unroll
      for (int q = 0; q < 8; ++q) {
        a[q]  = a_lo[q];  a[q + 8]  = a_hi[q];
        b0[q] = b0_lo[q]; b0[q + 8] = b0_hi[q];
        b1[q] = b1_lo[q]; b1[q + 8] = b1_hi[q];
      }
      acc0 = __builtin_amdgcn_wmma_f32_16x16x32_bf16(
          false, a, false, b0, (short)0, acc0, false, false);
      acc1 = __builtin_amdgcn_wmma_f32_16x16x32_bf16(
          false, a, false, b1, (short)0, acc1, false, false);

      wait_async_copies();
      __syncthreads();
    }

    // C/D layout: VGPR r -> M = r + 8*(lane>=16), N = lane&15.
    const int   mbase = wave * 16 + ((lane >> 4) << 3);
    const int   nc0   = n0 + (lane & 15);
    const int   nc1   = nc0 + 16;
    const float bv0   = bias[nc0];
    const float bv1   = bias[nc1];
#pragma unroll
    for (int r = 0; r < 8; ++r) {
      gates[(size_t)(mbase + r) * Gg + nc0] = acc0[r] + bv0;
      gates[(size_t)(mbase + r) * Gg + nc1] = acc1[r] + bv1;
    }

    grid_barrier(ctl);

    // ---------------- Phase B: LSTM cell elementwise -------------------------
    for (int e = blockIdx.x * TPB + tid; e < Bc * Hh; e += NWG * TPB) {
      const int b = e >> 9, j = e & 511;
      const float* gr = gates + (size_t)b * Gg;
      const float ig = sigmoid_f(gr[j]);
      const float fg = sigmoid_f(gr[j + 512]);
      const float cg = tanhf(gr[j + 1024]);
      const float og = sigmoid_f(gr[j + 1536]);
      const float cn = fg * cbuf[e] + ig * cg;
      const float hn = og * tanhf(cn);
      cbuf[e] = cn;
      out[((size_t)b * Tt + t) * Hh + j] = hn;
      hbf[e] = f2bf(hn);
      if (t == Tt - 1) { hy[e] = hn; cy[e] = cn; }
    }

    grid_barrier(ctl);
  }
}

// ------------------------------- launcher -----------------------------------
extern "C" void kernel_launch(void* const* d_in, const int* in_sizes, int n_in,
                              void* d_out, int out_size, void* d_ws, size_t ws_size,
                              hipStream_t stream) {
  (void)in_sizes; (void)n_in; (void)out_size; (void)ws_size;

  const float* x  = (const float*)d_in[0];
  const float* h0 = (const float*)d_in[1];
  const float* c0 = (const float*)d_in[2];
  // d_in[3] = ctx (unused by reference forward)
  const float* Wi = (const float*)d_in[4];
  const float* bi = (const float*)d_in[5];
  const float* Wh = (const float*)d_in[6];
  const float* bh = (const float*)d_in[7];

  float* out = (float*)d_out;
  float* hy  = out + (size_t)Bc * Tt * Hh;
  float* cy  = hy + (size_t)Bc * Hh;

  char* ws = (char*)d_ws;
  unsigned short* xbf  = (unsigned short*)(ws + OFF_XBF);
  unsigned short* wt   = (unsigned short*)(ws + OFF_WT);
  float*          bias = (float*)(ws + OFF_BIAS);
  float*          gts  = (float*)(ws + OFF_GATES);
  unsigned short* hbf  = (unsigned short*)(ws + OFF_HBF);
  float*          cbuf = (float*)(ws + OFF_C);
  int*            ctl  = (int*)(ws + OFF_CTL);

  const size_t nx = (size_t)Bc * Tt * Ii;              // 2^26, /4 per thread
  k_cvt_x <<<(unsigned)(nx / (4 * 256)), 256, 0, stream>>>(x, xbf);
  k_prep_w<<<(unsigned)(((size_t)Gg * KK) / 256), 256, 0, stream>>>(Wi, Wh, wt);
  k_init  <<<(Bc * Hh) / 256, 256, 0, stream>>>(h0, c0, bi, bh, hbf, cbuf, bias, ctl);
  lstm_scan<<<NWG, TPB, 0, stream>>>(xbf, wt, bias, gts, hbf, cbuf, out, hy, cy, ctl);
}
